// MeshProcessorGruBlock_4552665334039
// MI455X (gfx1250) — compile-verified, matrix operand order checked by reading, not run on hardware
//
#include <hip/hip_runtime.h>

// ---------------------------------------------------------------------------
// MeshGraphNet GRU processor block for MI455X (gfx1250, wave32).
//   edge GRU (EH=1536) -> edge MLP -> scatter-add -> node GRU (NH=1024)
//   -> node MLP -> layernorm.
// GEMMs: v_wmma_f32_16x16x32_bf16.  Shared A tiles staged into LDS by the
// Tensor Data Mover (tensor_load_to_lds, TENSORcnt), double-buffered and
// bank-conflict-free via TDM pad insertion (row stride 132 dwords).
// ---------------------------------------------------------------------------

typedef __bf16 bf16;
typedef __attribute__((ext_vector_type(8)))  __bf16 v8bf;
typedef __attribute__((ext_vector_type(16))) __bf16 v16bf;
typedef __attribute__((ext_vector_type(8)))  float  v8f;
typedef __attribute__((ext_vector_type(4)))  unsigned int u32x4;
typedef __attribute__((ext_vector_type(8)))  int i32x8;
typedef __attribute__((ext_vector_type(4)))  int i32x4;

#define E_EDGES 61440
#define N_NODES 10242
#define N_PAD   10256   // next multiple of 16 >= N_NODES
#define DD      512
#define EH      1536
#define NH      1024

// LDS chunk geometry: 16 rows x 256 bf16, TDM pads 16B after every 512B
// -> row stride 528 bytes (132 dwords == 4 mod 64 banks: conflict-free b128).
#define CHUNK_K      256
#define CHUNK_ROWB   528
#define CHUNK_BYTES  (16 * CHUNK_ROWB)          // 8448 per matrix
#define BUF_BYTES    (2 * CHUNK_BYTES)          // catX + catH per buffer
#define SMEM_BYTES   (2 * BUF_BYTES)            // double buffered = 33792

// ---- WMMA helpers ---------------------------------------------------------

__device__ __forceinline__ v8f wmma_bf16(v16bf a, v16bf b, v8f c) {
    return __builtin_amdgcn_wmma_f32_16x16x32_bf16(false, a, false, b,
                                                   (short)0, c, false, false);
}

// One lane's share of a 16x32 bf16 A/B fragment (ISA 7.12.2):
// p -> &M[row][k0 + (lane>>4)*8]; reads 8 bf16 at p and 8 at p+16.
__device__ __forceinline__ v16bf frag_ld(const bf16* __restrict__ p) {
    v8bf lo = *(const v8bf*)(p);
    v8bf hi = *(const v8bf*)(p + 16);
    return __builtin_shufflevector(lo, hi,
        0,1,2,3,4,5,6,7,8,9,10,11,12,13,14,15);
}

__device__ __forceinline__ float sigmoidf_(float x) {
    return 1.0f / (1.0f + __expf(-x));
}

// ---- Tensor Data Mover: 2D tile (16 rows x 256 bf16) global -> LDS --------
// D# per CDNA5 ISA.  data_size=2B, pad_enable,
// pad_interval=128 dwords (val 6), pad_amount=4 dwords (val 3),
// tensor_dim0=tile_dim0=256, tensor_dim1=tile_dim1=16, dim0_stride=H.
__device__ __forceinline__ void tdm_load_chunk(const bf16* gsrc,
                                               unsigned lds_off,
                                               int strideElems) {
    unsigned long long ga = (unsigned long long)gsrc;
    u32x4 g0;
    g0[0] = 1u;                                          // count=1 (valid)
    g0[1] = lds_off;                                     // lds_addr
    g0[2] = (unsigned)(ga & 0xffffffffu);                // global_addr[31:0]
    g0[3] = (unsigned)((ga >> 32) & 0x01ffffffu)         // global_addr[56:32]
          | (2u << 30);                                  // type=2 (image)
    i32x8 g1;
    g1[0] = (1 << 16)                                    // data_size = 2 bytes
          | (1 << 20)                                    // pad_enable
          | (6 << 22)                                    // pad_interval=128dw
          | (3 << 25);                                   // pad_amount=4dw
    g1[1] = (int)((unsigned)CHUNK_K << 16);              // tensor_dim0 lo16
    g1[2] = (int)(16u << 16);                            // dim0 hi | tensor_dim1 lo16
    g1[3] = (int)((unsigned)CHUNK_K << 16);              // dim1 hi | tile_dim0
    g1[4] = 16;                                          // tile_dim1 | tile_dim2=0
    g1[5] = strideElems;                                 // tensor_dim0_stride lo32
    g1[6] = 0;                                           // stride hi | dim1_stride lo
    g1[7] = 0;
    i32x4 gz4 = {0, 0, 0, 0};
    i32x8 gz8 = {0, 0, 0, 0, 0, 0, 0, 0};
    __builtin_amdgcn_tensor_load_to_lds(g0, g1, gz4, gz4, gz8, 0);
}

// ---- small utility kernels ------------------------------------------------

__global__ void f32_to_bf16_kernel(const float* __restrict__ in,
                                   bf16* __restrict__ out, long long n) {
    long long i = (long long)blockIdx.x * blockDim.x + threadIdx.x;
    long long stride = (long long)gridDim.x * blockDim.x;
    for (; i < n; i += stride) out[i] = (bf16)in[i];
}

__global__ void zero_f32_kernel(float* __restrict__ p, long long n) {
    long long i = (long long)blockIdx.x * blockDim.x + threadIdx.x;
    long long stride = (long long)gridDim.x * blockDim.x;
    for (; i < n; i += stride) p[i] = 0.0f;
}

__global__ void pack_edge_cat_kernel(const float* __restrict__ efeat,
                                     const float* __restrict__ nfeat,
                                     const float* __restrict__ ehid,
                                     const float* __restrict__ nhid,
                                     const int* __restrict__ src,
                                     const int* __restrict__ dst,
                                     bf16* __restrict__ catX,
                                     bf16* __restrict__ catH) {
    int e = blockIdx.x;
    int s = src[e], d = dst[e];
    bf16* cx = catX + (size_t)e * EH;
    bf16* ch = catH + (size_t)e * EH;
    const float* ef = efeat + (size_t)e * DD;
    const float* ns = nfeat + (size_t)s * DD;
    const float* nd = nfeat + (size_t)d * DD;
    const float* eh = ehid  + (size_t)e * DD;
    const float* hs = nhid  + (size_t)s * DD;
    const float* hd = nhid  + (size_t)d * DD;
    for (int c = threadIdx.x; c < DD; c += blockDim.x) {
        cx[c]          = (bf16)ef[c];
        cx[DD + c]     = (bf16)ns[c];
        cx[2 * DD + c] = (bf16)nd[c];
        ch[c]          = (bf16)eh[c];
        ch[DD + c]     = (bf16)hs[c];
        ch[2 * DD + c] = (bf16)hd[c];
    }
}

__global__ void pack_node_cat_kernel(const float* __restrict__ agg,
                                     const float* __restrict__ nfeat,
                                     const float* __restrict__ nhid,
                                     bf16* __restrict__ catX,
                                     bf16* __restrict__ catH) {
    int r = blockIdx.x;
    bf16* cx = catX + (size_t)r * NH;
    bf16* ch = catH + (size_t)r * NH;
    if (r >= N_NODES) {
        for (int c = threadIdx.x; c < NH; c += blockDim.x) {
            cx[c] = (bf16)0.0f;
            ch[c] = (bf16)0.0f;
        }
        return;
    }
    const float* a = agg   + (size_t)r * DD;
    const float* f = nfeat + (size_t)r * DD;
    const float* h = nhid  + (size_t)r * DD;
    for (int c = threadIdx.x; c < DD; c += blockDim.x) {
        bf16 av = (bf16)a[c];
        cx[c]      = av;
        ch[c]      = av;
        cx[DD + c] = (bf16)f[c];
        ch[DD + c] = (bf16)h[c];
    }
}

__global__ void scatter_add_kernel(const float* __restrict__ msg,
                                   const int* __restrict__ dst,
                                   float* __restrict__ agg) {
    int e = blockIdx.x;
    int d = dst[e];
    const float* m = msg + (size_t)e * DD;
    float* a = agg + (size_t)d * DD;
    for (int c = threadIdx.x; c < DD; c += blockDim.x)
        atomicAdd(a + c, m[c]);
}

__global__ void layernorm512_kernel(const float* __restrict__ x,
                                    const float* __restrict__ g,
                                    const float* __restrict__ b,
                                    float* __restrict__ out) {
    int r = blockIdx.x;
    const float* xr = x + (size_t)r * DD;
    float s = 0.0f, sq = 0.0f;
    for (int c = threadIdx.x; c < DD; c += blockDim.x) {
        float v = xr[c];
        s += v;
        sq += v * v;
    }
    __shared__ float sh1[256], sh2[256];
    sh1[threadIdx.x] = s;
    sh2[threadIdx.x] = sq;
    __syncthreads();
    for (int o = 128; o > 0; o >>= 1) {
        if ((int)threadIdx.x < o) {
            sh1[threadIdx.x] += sh1[threadIdx.x + o];
            sh2[threadIdx.x] += sh2[threadIdx.x + o];
        }
        __syncthreads();
    }
    float mu  = sh1[0] * (1.0f / DD);
    float var = sh2[0] * (1.0f / DD) - mu * mu;
    float inv = rsqrtf(var + 1e-5f);
    for (int c = threadIdx.x; c < DD; c += blockDim.x)
        out[(size_t)r * DD + c] = (xr[c] - mu) * inv * g[c] + b[c];
}

// ---- fused GRU GEMM kernel with TDM-staged A tiles ------------------------
// h' = GRU(catX, catH), hidden width H (K = H).  Wx/Wh bf16 [3H, H] (r,z,n).
// Grid: (rows/16, H/128); block 256 = 8 waves, one 16x16 C tile per wave.
// Wave 0 DMAs the shared 16-row A chunks (catX+catH, 16x256 bf16 each) into
// LDS via TDM; all waves consume them with conflict-free ds_load_b128.
__global__ void __launch_bounds__(256)
gru_wmma_kernel(const bf16* __restrict__ catX, const bf16* __restrict__ catH,
                const bf16* __restrict__ Wx, const float* __restrict__ bx,
                const bf16* __restrict__ Wh, const float* __restrict__ bh,
                bf16* __restrict__ hOut, int H) {
    __shared__ char smem[SMEM_BYTES];

    // LDS-store anchor: never executes (H is 1536 or 1024), but the compiler
    // cannot prove it, so it must treat smem as written -- this prevents the
    // "no stores to this LDS global -> loads are undef" fold that would
    // otherwise delete the ds_load consumption of TDM-written data.
    if (H & 0x80000000) smem[threadIdx.x] = (char)H;

    const int lane  = threadIdx.x & 31;
    const int wv    = threadIdx.x >> 5;            // wave id 0..7
    const int m0    = blockIdx.x << 4;
    const int n0    = (blockIdx.y << 7) + (wv << 4);
    const int row16 = lane & 15;
    const int koff  = (lane >> 4) << 3;            // 0 or 8
    const size_t Hs = (size_t)H;
    const size_t gate = Hs * Hs;

    const bf16* aXg = catX + (size_t)m0 * Hs;      // A tile base (uniform)
    const bf16* aHg = catH + (size_t)m0 * Hs;
    const bf16* bXr = Wx + (size_t)(n0 + row16) * Hs;
    const bf16* bXz = bXr + gate;
    const bf16* bXn = bXz + gate;
    const bf16* bHr = Wh + (size_t)(n0 + row16) * Hs;
    const bf16* bHz = bHr + gate;
    const bf16* bHn = bHz + gate;

    const int nChunks = H >> 8;                    // H / 256

    // Prologue: wave 0 DMAs chunk 0 of both matrices, waits, then barrier.
    if (threadIdx.x == 0) {
        tdm_load_chunk(aXg, 0u, H);
        tdm_load_chunk(aHg, (unsigned)CHUNK_BYTES, H);
        __builtin_amdgcn_s_wait_tensorcnt(0);
    }
    __syncthreads();

    v8f accR = {}, accZ = {}, accNX = {}, accNH = {};

    for (int kc = 0; kc < nChunks; ++kc) {
        // Prefetch next chunk into the other buffer (overlaps with compute).
        if (threadIdx.x == 0 && (kc + 1) < nChunks) {
            const int knext = (kc + 1) << 8;
            const unsigned bb = (unsigned)(((kc + 1) & 1) * BUF_BYTES);
            tdm_load_chunk(aXg + knext, bb, H);
            tdm_load_chunk(aHg + knext, bb + CHUNK_BYTES, H);
        }

        const char* xRow = smem + (kc & 1) * BUF_BYTES + row16 * CHUNK_ROWB;
        const char* hRow = xRow + CHUNK_BYTES;
        const int kbase = kc << 8;

#pragma unroll
        for (int k = 0; k < CHUNK_K; k += 32) {
            const int kk = k + koff;               // within chunk (<= 232)
            const int kg = kbase + kk;             // global K index
            // Issue all loads first so many stay in flight, then the WMMAs.
            v16bf ax = frag_ld((const bf16*)xRow + kk);   // LDS
            v16bf ah = frag_ld((const bf16*)hRow + kk);   // LDS
            v16bf b0 = frag_ld(bXr + kg);                 // global (L2 weights)
            v16bf b1 = frag_ld(bXz + kg);
            v16bf b2 = frag_ld(bXn + kg);
            v16bf b3 = frag_ld(bHr + kg);
            v16bf b4 = frag_ld(bHz + kg);
            v16bf b5 = frag_ld(bHn + kg);
            accR  = wmma_bf16(ax, b0, accR);
            accZ  = wmma_bf16(ax, b1, accZ);
            accNX = wmma_bf16(ax, b2, accNX);
            accR  = wmma_bf16(ah, b3, accR);
            accZ  = wmma_bf16(ah, b4, accZ);
            accNH = wmma_bf16(ah, b5, accNH);
        }

        // Ensure the prefetched buffer is resident before next iteration.
        if (threadIdx.x == 0) __builtin_amdgcn_s_wait_tensorcnt(0);
        __syncthreads();
    }

    // Epilogue: C layout -> m = m0 + 8*(lane>>4) + v, n = n0 + (lane&15)
    const int n  = n0 + row16;
    const int hi = lane >> 4;
    const float br   = bx[n]       + bh[n];
    const float bz   = bx[H + n]   + bh[H + n];
    const float bxn_ = bx[2 * H + n];
    const float bhn_ = bh[2 * H + n];
#pragma unroll
    for (int v = 0; v < 8; ++v) {
        const int m = m0 + hi * 8 + v;
        float r  = sigmoidf_(accR[v] + br);
        float z  = sigmoidf_(accZ[v] + bz);
        float nn = tanhf(accNX[v] + bxn_ + r * (accNH[v] + bhn_));
        float h  = (float)catH[(size_t)m * Hs + n];
        hOut[(size_t)m * Hs + n] = (bf16)((1.0f - z) * nn + z * h);
    }
}

// out[rows, 512] = A[rows, K](bf16) @ W[512, K]^T(bf16) + b  (fp32 out)
__global__ void __launch_bounds__(256)
mlp_wmma_kernel(const bf16* __restrict__ A, const bf16* __restrict__ W,
                const float* __restrict__ b, float* __restrict__ out, int K) {
    const int lane  = threadIdx.x & 31;
    const int wv    = threadIdx.x >> 5;
    const int m0    = blockIdx.x << 4;
    const int n0    = (blockIdx.y << 7) + (wv << 4);
    const int row16 = lane & 15;
    const int koff  = (lane >> 4) << 3;
    const size_t Ks = (size_t)K;

    const bf16* Ap = A + (size_t)(m0 + row16) * Ks;
    const bf16* Wp = W + (size_t)(n0 + row16) * Ks;

    v8f acc = {};
    for (int k = 0; k < K; k += 32) {
        const int kk = k + koff;
        acc = wmma_bf16(frag_ld(Ap + kk), frag_ld(Wp + kk), acc);
    }

    const int n = n0 + row16;
    const int hi = lane >> 4;
    const float bias = b[n];
#pragma unroll
    for (int v = 0; v < 8; ++v)
        out[(size_t)(m0 + hi * 8 + v) * DD + n] = acc[v] + bias;
}

// ---------------------------------------------------------------------------

static inline size_t align_up(size_t x, size_t a) { return (x + a - 1) & ~(a - 1); }

extern "C" void kernel_launch(void* const* d_in, const int* in_sizes, int n_in,
                              void* d_out, int out_size, void* d_ws, size_t ws_size,
                              hipStream_t stream) {
    const float* efeat   = (const float*)d_in[0];
    const float* nfeat   = (const float*)d_in[1];
    const float* ehidden = (const float*)d_in[2];
    const float* nhidden = (const float*)d_in[3];
    const int*   src     = (const int*)d_in[4];
    const int*   dst     = (const int*)d_in[5];
    const float* x2h_e_w = (const float*)d_in[6];
    const float* x2h_e_b = (const float*)d_in[7];
    const float* h2h_e_w = (const float*)d_in[8];
    const float* h2h_e_b = (const float*)d_in[9];
    const float* e_mlp_w = (const float*)d_in[10];
    const float* e_mlp_b = (const float*)d_in[11];
    const float* x2h_n_w = (const float*)d_in[12];
    const float* x2h_n_b = (const float*)d_in[13];
    const float* h2h_n_w = (const float*)d_in[14];
    const float* h2h_n_b = (const float*)d_in[15];
    const float* n_mlp_w = (const float*)d_in[16];
    const float* n_mlp_b = (const float*)d_in[17];
    const float* e_ln_g  = (const float*)d_in[18];
    const float* e_ln_b  = (const float*)d_in[19];
    const float* n_ln_g  = (const float*)d_in[20];
    const float* n_ln_b  = (const float*)d_in[21];

    float* out_e = (float*)d_out;                        // [E, 512]
    float* out_n = (float*)d_out + (size_t)E_EDGES * DD; // [N, 512]

    char* w = (char*)d_ws;
    size_t off = 0;
    auto carve = [&](size_t bytes) -> void* {
        void* p = w + off;
        off = align_up(off + bytes, 256);
        return p;
    };
    bf16* catXe    = (bf16*)carve((size_t)E_EDGES * EH * sizeof(bf16));
    bf16* catHe    = (bf16*)carve((size_t)E_EDGES * EH * sizeof(bf16));
    bf16* heNew    = (bf16*)carve((size_t)E_EDGES * EH * sizeof(bf16));
    bf16* WxeB     = (bf16*)carve((size_t)3 * EH * EH * sizeof(bf16));
    bf16* WheB     = (bf16*)carve((size_t)3 * EH * EH * sizeof(bf16));
    bf16* eMlpB    = (bf16*)carve((size_t)DD * EH * sizeof(bf16));
    bf16* WxnB     = (bf16*)carve((size_t)3 * NH * NH * sizeof(bf16));
    bf16* WhnB     = (bf16*)carve((size_t)3 * NH * NH * sizeof(bf16));
    bf16* nMlpB    = (bf16*)carve((size_t)DD * NH * sizeof(bf16));
    float* ehidNew = (float*)carve((size_t)E_EDGES * DD * sizeof(float));
    float* agg     = (float*)carve((size_t)N_NODES * DD * sizeof(float));
    bf16* catXn    = (bf16*)carve((size_t)N_PAD * NH * sizeof(bf16));
    bf16* catHn    = (bf16*)carve((size_t)N_PAD * NH * sizeof(bf16));
    bf16* hnNew    = (bf16*)carve((size_t)N_PAD * NH * sizeof(bf16));
    float* nhidNew = (float*)carve((size_t)N_PAD * DD * sizeof(float));
    (void)ws_size; (void)n_in; (void)in_sizes; (void)out_size;

    // 1. weights -> bf16
    auto cvt = [&](const float* s, bf16* d, long long n) {
        int blocks = (int)((n + 256LL * 8 - 1) / (256LL * 8));
        f32_to_bf16_kernel<<<blocks, 256, 0, stream>>>(s, d, n);
    };
    cvt(x2h_e_w, WxeB, 3LL * EH * EH);
    cvt(h2h_e_w, WheB, 3LL * EH * EH);
    cvt(e_mlp_w, eMlpB, (long long)DD * EH);
    cvt(x2h_n_w, WxnB, 3LL * NH * NH);
    cvt(h2h_n_w, WhnB, 3LL * NH * NH);
    cvt(n_mlp_w, nMlpB, (long long)DD * NH);

    // 2. gather + pack edge concat features (bf16)
    pack_edge_cat_kernel<<<E_EDGES, 256, 0, stream>>>(
        efeat, nfeat, ehidden, nhidden, src, dst, catXe, catHe);

    // 3. edge GRU (fused 6-GEMM WMMA, TDM-staged A)
    gru_wmma_kernel<<<dim3(E_EDGES / 16, EH / 128), 256, 0, stream>>>(
        catXe, catHe, WxeB, x2h_e_b, WheB, h2h_e_b, heNew, EH);

    // 4. edge MLP GEMM -> fp32 ehidden_new
    mlp_wmma_kernel<<<dim3(E_EDGES / 16, 4), 256, 0, stream>>>(
        heNew, eMlpB, e_mlp_b, ehidNew, EH);

    // 5. segment-sum messages at destination
    zero_f32_kernel<<<2048, 256, 0, stream>>>(agg, (long long)N_NODES * DD);
    scatter_add_kernel<<<E_EDGES, 256, 0, stream>>>(ehidNew, dst, agg);

    // 6. pack node concat features (padded to N_PAD rows)
    pack_node_cat_kernel<<<N_PAD, 256, 0, stream>>>(agg, nfeat, nhidden,
                                                    catXn, catHn);

    // 7. node GRU
    gru_wmma_kernel<<<dim3(N_PAD / 16, NH / 128), 256, 0, stream>>>(
        catXn, catHn, WxnB, x2h_n_b, WhnB, h2h_n_b, hnNew, NH);

    // 8. node MLP GEMM
    mlp_wmma_kernel<<<dim3(N_PAD / 16, 4), 256, 0, stream>>>(
        hnNew, nMlpB, n_mlp_b, nhidNew, NH);

    // 9. layernorms -> outputs
    layernorm512_kernel<<<E_EDGES, 256, 0, stream>>>(ehidNew, e_ln_g, e_ln_b, out_e);
    layernorm512_kernel<<<N_NODES, 256, 0, stream>>>(nhidNew, n_ln_g, n_ln_b, out_n);
}
